// AttentionHead1_77008763617383
// MI455X (gfx1250) — matrix operand order repaired and checked by reading, not compiled, not run on hardware
//
#include <hip/hip_runtime.h>

typedef __attribute__((ext_vector_type(16))) __bf16 v16bf;
typedef __attribute__((ext_vector_type(8)))  float  v8f;
typedef __attribute__((ext_vector_type(4)))  int    i32x4;

union BF16Frag { v16bf v; unsigned int u[8]; unsigned short h[16]; };

__device__ __forceinline__ unsigned short f2bf(float f) {
  unsigned int u = __builtin_bit_cast(unsigned int, f);
  u += 0x7FFFu + ((u >> 16) & 1u);           // round-to-nearest-even
  return (unsigned short)(u >> 16);
}
__device__ __forceinline__ unsigned int pack_bf2(float a, float b) {
  return (unsigned int)f2bf(a) | ((unsigned int)f2bf(b) << 16);
}

// -------- async global->LDS copy (CDNA5), with portable fallback ----------
#if defined(__has_builtin)
#  if __has_builtin(__builtin_amdgcn_global_load_async_to_lds_b128)
#    define HAVE_ASYNC_LDS 1
#  endif
#  if __has_builtin(__builtin_amdgcn_s_wait_asynccnt)
#    define HAVE_WAIT_ASYNC 1
#  endif
#endif
#ifndef HAVE_ASYNC_LDS
#  define HAVE_ASYNC_LDS 0
#endif
#ifndef HAVE_WAIT_ASYNC
#  define HAVE_WAIT_ASYNC 0
#endif

__device__ __forceinline__ void wait_asynccnt0() {
#if HAVE_WAIT_ASYNC
  __builtin_amdgcn_s_wait_asynccnt(0);
#else
  asm volatile("s_wait_asynccnt 0x0" ::: "memory");
#endif
}

#if HAVE_ASYNC_LDS
__device__ __forceinline__ void async_cp16(const void* g, void* l) {
  __builtin_amdgcn_global_load_async_to_lds_b128(
      (__attribute__((address_space(1))) i32x4*)g,
      (__attribute__((address_space(3))) i32x4*)l, 0, 0);
}
#endif

// ---------------------------------------------------------------------------
// Kernel 0: one-shot weight prepack into B-fragment lane layout (bf16 pairs).
// Index: (((m*24 + ks)*4 + cbi)*32 + lane)*8 + v
//   element pair = W[(ks*32 + (lane<16?0:16) + 2v + {0,1}) * 64 + cbi*16 + lane%16]
// ---------------------------------------------------------------------------
__global__ __launch_bounds__(256) void pack_w(
    const float* __restrict__ Wq, const float* __restrict__ Wk,
    const float* __restrict__ Wv, unsigned int* __restrict__ pk)
{
  const int idx = blockIdx.x * 256 + threadIdx.x;    // 3*24*4*32*8 = 73728
  const int v    = idx & 7;
  const int lane = (idx >> 3) & 31;
  const int cbi  = (idx >> 8) & 3;
  const int rest = idx >> 10;                        // m*24 + ks  (0..71)
  const int ks   = rest % 24;
  const int m    = rest / 24;
  const float* W = (m == 0) ? Wq : ((m == 1) ? Wk : Wv);
  const int krow = ks * 32 + ((lane & 16) ? 16 : 0) + 2 * v;
  const int col  = cbi * 16 + (lane & 15);
  pk[idx] = pack_bf2(W[krow * 64 + col], W[(krow + 1) * 64 + col]);
}

// ---------------------------------------------------------------------------
// Kernel 1: fused QKV projection.  X[16 x 768] staged in LDS as bf16.
// 4 waves/block; each wave owns 3 of 12 output 16-col tiles (Q|K|V).
// Weights come prepacked: two global_load_b128 per fragment.
// ---------------------------------------------------------------------------
__global__ __launch_bounds__(128) void qkv_proj(
    const float* __restrict__ X,
    const float* __restrict__ bq, const float* __restrict__ bk,
    const float* __restrict__ bv, const unsigned int* __restrict__ pk,
    unsigned short* __restrict__ Qb, unsigned short* __restrict__ Kb,
    unsigned short* __restrict__ Vb)
{
  constexpr int DIN = 768, DOUT = 64;
  constexpr float QSCALE = 0.022097086912079608f;   // 1/sqrt(2048)
  __shared__ unsigned short sX[16 * DIN];

  const int tid  = threadIdx.x;
  const int lane = tid & 31;
  const int wave = tid >> 5;
  const int l16  = lane & 15;
  const int hi   = (lane >= 16);
  const long row0 = (long)blockIdx.x * 16;          // flat over B*S

  for (int i = tid; i < 16 * DIN; i += 128) {
    int r = i / DIN, c = i - r * DIN;
    sX[i] = f2bf(X[(row0 + r) * DIN + c]);
  }
  __syncthreads();

  v8f zero = {};
  v8f acc0 = zero, acc1 = zero, acc2 = zero;

  const float* bt[3]; unsigned short* Ot[3];
  int cb[3], mm[3]; long woff[3];
#pragma unroll
  for (int j = 0; j < 3; ++j) {
    int t = wave * 3 + j;                 // tile 0..11
    int m = t >> 2;                       // 0=Q 1=K 2=V
    mm[j] = m;
    cb[j] = (t & 3) * 16;
    bt[j] = (m == 0) ? bq : ((m == 1) ? bk : bv);
    Ot[j] = (m == 0) ? Qb : ((m == 1) ? Kb : Vb);
    woff[j] = (long)m * 24 * 1024 + (long)(t & 3) * 256 + lane * 8;
  }

  for (int ks = 0; ks < DIN / 32; ++ks) {
    const int k0 = ks * 32;
    BF16Frag a;
    {
      const unsigned short* rp = &sX[l16 * DIN];
      int kb0 = k0 + (hi ? 16 : 0);
      int kb1 = k0 + (hi ? 24 : 8);
      *(uint4*)&a.u[0] = *(const uint4*)(rp + kb0);   // ds_load_b128
      *(uint4*)&a.u[4] = *(const uint4*)(rp + kb1);
    }
#pragma unroll
    for (int j = 0; j < 3; ++j) {
      BF16Frag bf;
      const unsigned int* wp = pk + woff[j] + (long)ks * 1024;
      *(uint4*)&bf.u[0] = *(const uint4*)(wp);        // global_load_b128
      *(uint4*)&bf.u[4] = *(const uint4*)(wp + 4);
      v8f& acc = (j == 0) ? acc0 : ((j == 1) ? acc1 : acc2);
      acc = __builtin_amdgcn_wmma_f32_16x16x32_bf16(
          false, a.v, false, bf.v, (short)0, acc, false, false);
    }
  }

#pragma unroll
  for (int j = 0; j < 3; ++j) {
    const int n = cb[j] + l16;
    const float bias = bt[j][n];
    const float sc = (mm[j] == 0) ? QSCALE : 1.0f;
    unsigned short* op = Ot[j];
    v8f acc = (j == 0) ? acc0 : ((j == 1) ? acc1 : acc2);
#pragma unroll
    for (int r = 0; r < 8; ++r) {
      int M = r + (hi ? 8 : 0);
      op[(row0 + M) * DOUT + n] = f2bf((acc[r] + bias) * sc);
    }
  }
}

// ---------------------------------------------------------------------------
// Kernel 2: flash attention.  One block = one batch x 128 query rows
// (8 waves x 16-row tiles).  K/V blocks staged via async global->LDS copies,
// next block prefetched; scores never touch HBM.
// ---------------------------------------------------------------------------
__global__ __launch_bounds__(256) void flash_attn(
    const unsigned short* __restrict__ Qb, const unsigned short* __restrict__ Kb,
    const unsigned short* __restrict__ Vb, const unsigned char* __restrict__ mask,
    float* __restrict__ out)
{
  constexpr int S = 2048, D = 64;
  __shared__ unsigned short sK[32 * D];       // [key][dim]  bf16
  __shared__ unsigned short sV[32 * D];       // [key][dim]  bf16
  __shared__ unsigned short sP[8][16 * 32];   // per-wave P tile

  const int tid  = threadIdx.x;
  const int lane = tid & 31;
  const int wave = tid >> 5;
  const int l16  = lane & 15;
  const int hi   = (lane >= 16);

  const int b  = blockIdx.x >> 4;             // batch
  const int q0 = (blockIdx.x & 15) * 128 + wave * 16;

  // Q A-fragments: dims 0..31 and 32..63 (1/sqrt(S) already folded in).
  BF16Frag qa0, qa1;
  {
    const unsigned short* qp = Qb + ((long)b * S + q0 + l16) * D;
    const int kb0 = hi ? 16 : 0, kb1 = hi ? 24 : 8;
    *(uint4*)&qa0.u[0] = *(const uint4*)(qp + kb0);
    *(uint4*)&qa0.u[4] = *(const uint4*)(qp + kb1);
    *(uint4*)&qa1.u[0] = *(const uint4*)(qp + 32 + kb0);
    *(uint4*)&qa1.u[4] = *(const uint4*)(qp + 32 + kb1);
  }

  v8f zero = {};
  v8f o0 = zero, o1 = zero, o2 = zero, o3 = zero;
  float mrun[8], lrun[8];
#pragma unroll
  for (int r = 0; r < 8; ++r) { mrun[r] = -3.0e38f; lrun[r] = 0.0f; }

  const unsigned char* mp = mask + (long)b * S * S;
  unsigned short* sPw = &sP[wave][0];

  for (int kb = 0; kb < S / 32; ++kb) {
    __syncthreads();                           // prior-iteration readers done
    {
      const long base = ((long)b * S + kb * 32) * D;   // ushort index
#if HAVE_ASYNC_LDS
      const char* gK = (const char*)(Kb + base);
      const char* gV = (const char*)(Vb + base);
      char* lK = (char*)sK;
      char* lV = (char*)sV;
      async_cp16(gK + tid * 16, lK + tid * 16);   // global_load_async_to_lds_b128
      async_cp16(gV + tid * 16, lV + tid * 16);
#else
      for (int i = tid; i < 32 * D; i += 256) {
        sK[i] = Kb[base + i];
        sV[i] = Vb[base + i];
      }
#endif
      // prefetch next K/V block while this one lands / computes
      if (kb + 1 < S / 32) {
        const char* nK = (const char*)(Kb + base + 32 * D);
        const char* nV = (const char*)(Vb + base + 32 * D);
        __builtin_prefetch(nK + tid * 16, 0, 0);   // global_prefetch_b8
        __builtin_prefetch(nV + tid * 16, 0, 0);
      }
#if HAVE_ASYNC_LDS
      wait_asynccnt0();
#endif
    }
    __syncthreads();

    // ---- scores: 16 queries x 32 keys, dot over D=64 (2 chained WMMAs) ----
    v8f s0 = zero, s1 = zero;
    {
      const int kdim0 = hi ? 16 : 0;
      BF16Frag kf;
      const unsigned short* kp0 = &sK[l16 * D];          // keys 0..15
      *(uint4*)&kf.u[0] = *(const uint4*)(kp0 + kdim0);
      *(uint4*)&kf.u[4] = *(const uint4*)(kp0 + kdim0 + 8);
      s0 = __builtin_amdgcn_wmma_f32_16x16x32_bf16(false, qa0.v, false, kf.v, (short)0, s0, false, false);
      *(uint4*)&kf.u[0] = *(const uint4*)(kp0 + 32 + kdim0);
      *(uint4*)&kf.u[4] = *(const uint4*)(kp0 + 32 + kdim0 + 8);
      s0 = __builtin_amdgcn_wmma_f32_16x16x32_bf16(false, qa1.v, false, kf.v, (short)0, s0, false, false);
      const unsigned short* kp1 = &sK[(16 + l16) * D];   // keys 16..31
      *(uint4*)&kf.u[0] = *(const uint4*)(kp1 + kdim0);
      *(uint4*)&kf.u[4] = *(const uint4*)(kp1 + kdim0 + 8);
      s1 = __builtin_amdgcn_wmma_f32_16x16x32_bf16(false, qa0.v, false, kf.v, (short)0, s1, false, false);
      *(uint4*)&kf.u[0] = *(const uint4*)(kp1 + 32 + kdim0);
      *(uint4*)&kf.u[4] = *(const uint4*)(kp1 + 32 + kdim0 + 8);
      s1 = __builtin_amdgcn_wmma_f32_16x16x32_bf16(false, qa1.v, false, kf.v, (short)0, s1, false, false);
    }

    // ---- mask + online softmax (16-lane butterflies per row group) ----
    float p0[8], p1[8];
    {
      const long mrow0 = (long)(q0 + (hi ? 8 : 0));
      const int kc0 = kb * 32 + l16;
#pragma unroll
      for (int r = 0; r < 8; ++r) {
        const unsigned char* mr = mp + (mrow0 + r) * S;
        float a0 = s0[r], a1 = s1[r];
        a0 = mr[kc0]      ? -1.0e9f : a0;
        a1 = mr[kc0 + 16] ? -1.0e9f : a1;
        float rmax = fmaxf(a0, a1);
        rmax = fmaxf(rmax, __shfl_xor(rmax, 1, 32));
        rmax = fmaxf(rmax, __shfl_xor(rmax, 2, 32));
        rmax = fmaxf(rmax, __shfl_xor(rmax, 4, 32));
        rmax = fmaxf(rmax, __shfl_xor(rmax, 8, 32));
        float mnew = fmaxf(mrun[r], rmax);
        float e0 = __expf(a0 - mnew);
        float e1 = __expf(a1 - mnew);
        float rs = e0 + e1;
        rs += __shfl_xor(rs, 1, 32);
        rs += __shfl_xor(rs, 2, 32);
        rs += __shfl_xor(rs, 4, 32);
        rs += __shfl_xor(rs, 8, 32);
        float sf = __expf(mrun[r] - mnew);
        lrun[r] = lrun[r] * sf + rs;
        mrun[r] = mnew;
        o0[r] *= sf; o1[r] *= sf; o2[r] *= sf; o3[r] *= sf;
        p0[r] = e0; p1[r] = e1;
      }
    }

    // ---- P (C-layout) -> LDS -> A-layout fragment (wave-local) ----
#pragma unroll
    for (int r = 0; r < 8; ++r) {
      int M = r + (hi ? 8 : 0);
      sPw[M * 32 + l16]      = f2bf(p0[r]);
      sPw[M * 32 + 16 + l16] = f2bf(p1[r]);
    }
    asm volatile("s_wait_dscnt 0x0" ::: "memory");

    BF16Frag pa;
    {
      const unsigned short* pp = &sPw[l16 * 32];
      const int pk0 = hi ? 16 : 0, pk1 = hi ? 24 : 8;
      *(uint4*)&pa.u[0] = *(const uint4*)(pp + pk0);
      *(uint4*)&pa.u[4] = *(const uint4*)(pp + pk1);
    }

    // ---- O += P x V  (4 column tiles of 16; V gathered from row-major) ----
    const int vk0 = hi ? 16 : 0;
#pragma unroll
    for (int a = 0; a < 4; ++a) {
      BF16Frag vf;
      const int dim = a * 16 + l16;
#pragma unroll
      for (int v = 0; v < 8; ++v) {
        unsigned int h0 = sV[(vk0 + 2 * v)     * D + dim];
        unsigned int h1 = sV[(vk0 + 2 * v + 1) * D + dim];
        vf.u[v] = h0 | (h1 << 16);
      }
      v8f& oa = (a == 0) ? o0 : ((a == 1) ? o1 : ((a == 2) ? o2 : o3));
      oa = __builtin_amdgcn_wmma_f32_16x16x32_bf16(
          false, pa.v, false, vf.v, (short)0, oa, false, false);
    }
  }

  // ---- epilogue: normalize and store fp32 ----
#pragma unroll
  for (int r = 0; r < 8; ++r) {
    float inv = 1.0f / lrun[r];
    int M = r + (hi ? 8 : 0);
    float* orow = out + ((long)b * S + q0 + M) * D;
    orow[l16]      = o0[r] * inv;
    orow[16 + l16] = o1[r] * inv;
    orow[32 + l16] = o2[r] * inv;
    orow[48 + l16] = o3[r] * inv;
  }
}

// ---------------------------------------------------------------------------
extern "C" void kernel_launch(void* const* d_in, const int* in_sizes, int n_in,
                              void* d_out, int out_size, void* d_ws, size_t ws_size,
                              hipStream_t stream) {
  (void)in_sizes; (void)n_in; (void)out_size; (void)ws_size;
  const float*         X    = (const float*)d_in[0];
  const unsigned char* mask = (const unsigned char*)d_in[1];
  const float* Wq = (const float*)d_in[2];
  const float* bq = (const float*)d_in[3];
  const float* Wk = (const float*)d_in[4];
  const float* bk = (const float*)d_in[5];
  const float* Wv = (const float*)d_in[6];
  const float* bv = (const float*)d_in[7];
  float* out = (float*)d_out;

  constexpr long BS = 8L * 2048;
  unsigned short* Qb = (unsigned short*)d_ws;          // 3 x BS x 64 bf16 = 6 MB
  unsigned short* Kb = Qb + BS * 64;
  unsigned short* Vb = Kb + BS * 64;
  unsigned int*   pk = (unsigned int*)(Vb + BS * 64);  // 73728 u32 = 288 KB

  pack_w<<<dim3(288), dim3(256), 0, stream>>>(Wq, Wk, Wv, pk);
  qkv_proj<<<dim3((unsigned)(BS / 16)), dim3(128), 0, stream>>>(
      X, bq, bk, bv, pk, Qb, Kb, Vb);
  flash_attn<<<dim3(8 * (2048 / 128)), dim3(256), 0, stream>>>(
      Qb, Kb, Vb, mask, out);
}
